// BasicGCN_83227876262380
// MI455X (gfx1250) — matrix operand order repaired and checked by reading, not compiled, not run on hardware
//
#include <hip/hip_runtime.h>
#include <math.h>

typedef __attribute__((ext_vector_type(16))) _Float16 v16h;
typedef __attribute__((ext_vector_type(8)))  float    v8f;

// ---------------------------------------------------------------------------
// WMMA fragment loaders (CDNA5 wave32 layouts, ISA 7.12.2)
// ---------------------------------------------------------------------------

// A: 16x32 f16 tile from f32 matrix A (row-major, leading dim lda), rows
// row0..row0+15 (optionally gathered through an index array), cols kOff..kOff+31.
// Lanes 0-15: M=lane, K in {0..7, 16..23}; lanes 16-31: M=lane-16, K in {8..15, 24..31}.
__device__ __forceinline__ v16h load_a_frag(const float* __restrict__ A, int lda,
                                            const int* __restrict__ gather,
                                            int row0, int kOff, int lane) {
  int m  = lane & 15;
  int kb = (lane < 16) ? 0 : 8;
  int r  = gather ? gather[row0 + m] : (row0 + m);
  const float* p = A + (size_t)r * lda + kOff + kb;
  v16h a;
#pragma unroll
  for (int i = 0; i < 8; ++i)  a[i] = (_Float16)p[i];        // K = kb + i
#pragma unroll
  for (int i = 8; i < 16; ++i) a[i] = (_Float16)p[i + 8];    // K = kb + 16 + (i-8)
  return a;
}

// B: 32x16 f16 tile from f32 weight matrix W (row-major K x Nw, leading dim ldw),
// rows kOff..kOff+31, cols n0..n0+15.
// Lanes 0-15: N=lane, K in {0..7, 16..23}; lanes 16-31: N=lane-16, K in {8..15, 24..31}.
__device__ __forceinline__ v16h load_b_frag(const float* __restrict__ W, int ldw,
                                            int n0, int kOff, int lane) {
  int n  = lane & 15;
  int kb = (lane < 16) ? 0 : 8;
  const float* p = W + (size_t)(kOff + kb) * ldw + n0 + n;
  v16h b;
#pragma unroll
  for (int i = 0; i < 16; ++i) b[i] = (_Float16)p[(size_t)(i < 8 ? i : i + 8) * ldw];
  return b;
}

// D/C: 16x16 f32. Lane L holds column (L&15); VGPR j holds row 8*(L>=16)+j.
__device__ __forceinline__ void store_d_bias_relu(float* __restrict__ O, int ldo,
                                                  int row0, int n0,
                                                  const float* __restrict__ bias,
                                                  v8f c, int lane) {
  int n  = lane & 15;
  int mb = (lane < 16) ? 0 : 8;
  float bv = bias[n0 + n];
#pragma unroll
  for (int j = 0; j < 8; ++j) {
    float v = c[j] + bv;
    v = fmaxf(v, 0.0f);
    O[(size_t)(row0 + mb + j) * ldo + n0 + n] = v;
  }
}

// ---------------------------------------------------------------------------
// Generic fused (gather) GEMM + bias + ReLU:  O = relu(A[idx] * W + b)
// One wave computes a 16 x (NT*16) output tile; KC chunks of K=32.
// ---------------------------------------------------------------------------
template <int KC, int NT>
__global__ void gemm_bias_relu(const float* __restrict__ A, int lda,
                               const int* __restrict__ gather,
                               const float* __restrict__ W, int ldw,
                               const float* __restrict__ bias,
                               float* __restrict__ O, int ldo,
                               int Mtiles) {
  int wave = blockIdx.x * (blockDim.x >> 5) + (threadIdx.x >> 5);
  if (wave >= Mtiles) return;          // whole-wave exit; active waves keep EXEC all-1s
  int lane = threadIdx.x & 31;
  int row0 = wave * 16;

  v8f acc[NT];
#pragma unroll
  for (int t = 0; t < NT; ++t) acc[t] = (v8f){0.f,0.f,0.f,0.f,0.f,0.f,0.f,0.f};

#pragma unroll
  for (int kc = 0; kc < KC; ++kc) {
    v16h a = load_a_frag(A, lda, gather, row0, kc * 32, lane);
#pragma unroll
    for (int t = 0; t < NT; ++t) {
      v16h b = load_b_frag(W, ldw, t * 16, kc * 32, lane);
      acc[t] = __builtin_amdgcn_wmma_f32_16x16x32_f16(
          /*neg_a=*/false, a, /*neg_b=*/false, b,
          /*c_mod=*/(short)0, acc[t], /*reuse_a=*/false, /*reuse_b=*/false);
    }
  }
#pragma unroll
  for (int t = 0; t < NT; ++t)
    store_d_bias_relu(O, ldo, row0, t * 16, bias, acc[t], lane);
}

// ---------------------------------------------------------------------------
// SpMM: S[row[e], d] += val[e] * X[col[e], d]   (COO gather-scale-scatter)
// One lane per (edge, feature); the 32 feature lanes coalesce the X row read.
// ---------------------------------------------------------------------------
__global__ void spmm_scatter(const int* __restrict__ row, const int* __restrict__ col,
                             const float* __restrict__ val,
                             const float* __restrict__ X, int ldx,
                             float* __restrict__ S, long long nWork) {
  long long t = (long long)blockIdx.x * blockDim.x + threadIdx.x;
  if (t >= nWork) return;
  int e = (int)(t >> 5);
  int d = (int)(t & 31);
  float contrib = val[e] * X[(size_t)col[e] * ldx + d];
  unsafeAtomicAdd(&S[(size_t)row[e] * 32 + d], contrib);  // global_atomic_add_f32
}

__global__ void zero_f32(float* __restrict__ p, long long n) {
  long long t = (long long)blockIdx.x * blockDim.x + threadIdx.x;
  if (t < n) p[t] = 0.0f;
}

// Copy embeddings [N,32] into columns 0..31 of h [N,128]
__global__ void copy_emb(const float* __restrict__ emb, float* __restrict__ h,
                         long long n) {
  long long t = (long long)blockIdx.x * blockDim.x + threadIdx.x;
  if (t < n) h[(t >> 5) * 128 + (t & 31)] = emb[t];
}

// Final: out[r] = sigmoid(dot(z2[r,0:16], w) + b)
__global__ void clf_final(const float* __restrict__ Z, const float* __restrict__ w,
                          const float* __restrict__ b, float* __restrict__ out, int B) {
  int r = blockIdx.x * blockDim.x + threadIdx.x;
  if (r >= B) return;
  float s = b[0];
#pragma unroll
  for (int k = 0; k < 16; ++k) s += Z[(size_t)r * 16 + k] * w[k];
  out[r] = 1.0f / (1.0f + __expf(-s));
}

// ---------------------------------------------------------------------------
// Host launch
// ---------------------------------------------------------------------------
extern "C" void kernel_launch(void* const* d_in, const int* in_sizes, int n_in,
                              void* d_out, int out_size, void* d_ws, size_t ws_size,
                              hipStream_t stream) {
  const float* emb = (const float*)d_in[0];
  const float* W[3]  = {(const float*)d_in[1], (const float*)d_in[3], (const float*)d_in[5]};
  const float* bW[3] = {(const float*)d_in[2], (const float*)d_in[4], (const float*)d_in[6]};
  const float* unet_W0 = (const float*)d_in[7];
  const float* unet_b0 = (const float*)d_in[8];
  const float* unet_W1 = (const float*)d_in[9];
  const float* unet_b1 = (const float*)d_in[10];
  const float* inet_W0 = (const float*)d_in[11];
  const float* inet_b0 = (const float*)d_in[12];
  const float* inet_W1 = (const float*)d_in[13];
  const float* inet_b1 = (const float*)d_in[14];
  const float* clf_W0  = (const float*)d_in[15];
  const float* clf_b0  = (const float*)d_in[16];
  const float* clf_W1  = (const float*)d_in[17];
  const float* clf_b1  = (const float*)d_in[18];
  const float* clf_W2  = (const float*)d_in[19];
  const float* clf_b2  = (const float*)d_in[20];
  const int*   rowi = (const int*)d_in[21];
  const int*   coli = (const int*)d_in[22];
  const float* vali = (const float*)d_in[23];
  const int*   u    = (const int*)d_in[24];
  const int*   iidx = (const int*)d_in[25];

  const int N = in_sizes[0] / 32;   // 100000
  const int E = in_sizes[21];       // 1700000
  const int B = in_sizes[24];       // 16384

  // workspace carve-up (floats)
  float* h   = (float*)d_ws;                  // [N,128]
  float* s   = h   + (size_t)N * 128;         // [N,32] spmm scratch
  float* ufh = s   + (size_t)N * 32;          // [B,64]
  float* vfh = ufh + (size_t)B * 64;          // [B,64]
  float* zbf = vfh + (size_t)B * 64;          // [B,64] concat(uf, vf)
  float* z1  = zbf + (size_t)B * 64;          // [B,32]
  float* z2  = z1  + (size_t)B * 32;          // [B,16]

  const int TB = 256;
  auto blocks1d = [](long long n, int tb) { return (unsigned)((n + tb - 1) / tb); };

  // h[:, 0:32] = embeddings
  {
    long long n = (long long)N * 32;
    copy_emb<<<blocks1d(n, TB), TB, 0, stream>>>(emb, h, n);
  }

  // 3 GCN layers: s = A_hat @ x_l ; h[:, 32*(l+1):...] = relu(s @ W_l + b_l)
  const int MtN = N / 16;           // 6250 row tiles
  for (int l = 0; l < 3; ++l) {
    long long ns = (long long)N * 32;
    zero_f32<<<blocks1d(ns, TB), TB, 0, stream>>>(s, ns);
    long long nw = (long long)E * 32;
    spmm_scatter<<<blocks1d(nw, TB), TB, 0, stream>>>(
        rowi, coli, vali, h + 32 * l, 128, s, nw);
    gemm_bias_relu<1, 2><<<(MtN + 7) / 8, TB, 0, stream>>>(
        s, 32, nullptr, W[l], 32, bW[l], h + 32 * (l + 1), 128, MtN);
  }

  const int MtB = B / 16;           // 1024 row tiles

  // user tower: relu(relu(h[u] @ unet_W0 + b0) @ unet_W1 + b1) -> zbuf[:, 0:32]
  gemm_bias_relu<4, 4><<<(MtB + 7) / 8, TB, 0, stream>>>(
      h, 128, u, unet_W0, 64, unet_b0, ufh, 64, MtB);
  gemm_bias_relu<2, 2><<<(MtB + 7) / 8, TB, 0, stream>>>(
      ufh, 64, nullptr, unet_W1, 32, unet_b1, zbf, 64, MtB);

  // item tower -> zbuf[:, 32:64]
  gemm_bias_relu<4, 4><<<(MtB + 7) / 8, TB, 0, stream>>>(
      h, 128, iidx, inet_W0, 64, inet_b0, vfh, 64, MtB);
  gemm_bias_relu<2, 2><<<(MtB + 7) / 8, TB, 0, stream>>>(
      vfh, 64, nullptr, inet_W1, 32, inet_b1, zbf + 32, 64, MtB);

  // classifier head
  gemm_bias_relu<2, 2><<<(MtB + 7) / 8, TB, 0, stream>>>(
      zbf, 64, nullptr, clf_W0, 32, clf_b0, z1, 32, MtB);
  gemm_bias_relu<1, 1><<<(MtB + 7) / 8, TB, 0, stream>>>(
      z1, 32, nullptr, clf_W1, 16, clf_b1, z2, 16, MtB);
  clf_final<<<blocks1d(B, TB), TB, 0, stream>>>(z2, clf_W2, clf_b2, (float*)d_out, B);
}
